// XEyTransformerLayer_57019985822461
// MI455X (gfx1250) — matrix operand order repaired and checked by reading, not compiled, not run on hardware
//
#include <hip/hip_runtime.h>
#include <math.h>

// ---------------------------------------------------------------------------
// XEyTransformerLayer for MI455X (gfx1250, wave32, WMMA).
// Bandwidth-bound: fused float4 elementwise passes over the 37.7MB edge
// tensor; all linears via v_wmma_f32_16x16x32_f16 (f16 in, f32 accumulate)
// with guard-free b128 tile staging and b128 LDS fragment loads.
// ---------------------------------------------------------------------------

typedef __attribute__((ext_vector_type(16))) _Float16 v16h;
typedef __attribute__((ext_vector_type(8)))  _Float16 v8h;
typedef __attribute__((ext_vector_type(8)))  float    v8f;

#define BSZ 4
#define NN  192
#define DXC 256
#define DEC 64
#define DYC 64
#define NHC 8

// ===========================================================================
// Fast WMMA GEMM: requires M%32==0, N%64==0, K%32==0.
//   A element (m,k) at A[m*lda + k*inca]; B element (k,n) at B[k*ldb + n].
// Workgroup = 8 wave32s -> 32x64 tile. CONTIG: inca==1 (branch-free A b128).
// B tile staged TRANSPOSED in LDS so fragments are contiguous b128 loads.
// ===========================================================================
template <bool CONTIG>
__global__ __launch_bounds__(256) void xe_gemm_fast(
    const float* __restrict__ A, int lda, int inca,
    const float* __restrict__ B, int ldb,
    const float* __restrict__ bias,
    float* __restrict__ C, int ldc,
    int M, int K, int Nn, int act)
{
  __shared__ _Float16 As[32][32];            // [m][k]
  __shared__ _Float16 Bt[64][48];            // [n][k], row 96B for 16B align

  const int tid  = threadIdx.x;
  const int wave = tid >> 5;
  const int lane = tid & 31;
  const int rt   = wave >> 2;                // 0..1
  const int ct   = wave & 3;                 // 0..3
  const int half = lane >> 4;
  const int l16  = lane & 15;
  const int row0 = blockIdx.x * 32;
  const int col0 = blockIdx.y * 64;

  const int ar = tid >> 3, ac = (tid & 7) << 2;   // A: 4 floats / thread
  const int kr = (tid >> 4) << 1;                 // B: k-pair row (0,2,..,30)
  const int bc = (tid & 15) << 2;                 // B: 4 cols / thread

  v8f acc = {};

  for (int k0 = 0; k0 < K; k0 += 32) {
    // ---- A tile: global b128 load -> f16 -> ds b64 store ----
    {
      union { _Float16 h[4]; unsigned long long u; } pk;
      if (CONTIG) {
        const float4 f = *(const float4*)(A + (size_t)(row0 + ar) * lda + k0 + ac);
        pk.h[0] = (_Float16)f.x; pk.h[1] = (_Float16)f.y;
        pk.h[2] = (_Float16)f.z; pk.h[3] = (_Float16)f.w;
      } else {
        const float* ap = A + (size_t)(row0 + ar) * lda + (size_t)(k0 + ac) * inca;
        pk.h[0] = (_Float16)ap[0];
        pk.h[1] = (_Float16)ap[inca];
        pk.h[2] = (_Float16)ap[2 * inca];
        pk.h[3] = (_Float16)ap[3 * inca];
      }
      *(unsigned long long*)&As[ar][ac] = pk.u;
    }
    // ---- B tile: 2 k-rows x 4 cols -> 4 packed b32 stores (transposed) ----
    {
      const float* b0 = B + (size_t)(k0 + kr) * ldb + col0 + bc;
      const float4 f0 = *(const float4*)(b0);
      const float4 f1 = *(const float4*)(b0 + ldb);
      union { _Float16 h[2]; unsigned int u; } q0, q1, q2, q3;
      q0.h[0] = (_Float16)f0.x; q0.h[1] = (_Float16)f1.x;
      q1.h[0] = (_Float16)f0.y; q1.h[1] = (_Float16)f1.y;
      q2.h[0] = (_Float16)f0.z; q2.h[1] = (_Float16)f1.z;
      q3.h[0] = (_Float16)f0.w; q3.h[1] = (_Float16)f1.w;
      *(unsigned int*)&Bt[bc + 0][kr] = q0.u;
      *(unsigned int*)&Bt[bc + 1][kr] = q1.u;
      *(unsigned int*)&Bt[bc + 2][kr] = q2.u;
      *(unsigned int*)&Bt[bc + 3][kr] = q3.u;
    }
    __syncthreads();

    // speculative prefetch of next tiles (dropped safely if OOB)
    if (k0 + 32 < K) {
      __builtin_prefetch(A + (size_t)(row0 + ar) * lda +
                         (size_t)(k0 + 32 + ac) * (CONTIG ? 1 : inca), 0, 3);
      __builtin_prefetch(B + (size_t)(k0 + 32 + kr) * ldb + col0 + bc, 0, 3);
    }

    // Fragments per CDNA5 16-bit layouts (ISA 7.12.2):
    //  A 16x32: lane<16 -> row l16, K {0..7,16..23}; lane>=16 -> K {8..15,24..31}
    //  B 32x16: lane<16 -> col l16, K {0..7,16..23}; lane>=16 -> K {8..15,24..31}
    const int arow = rt * 16 + l16;
    const int kbA  = half * 8;
    const v8h alo = *(const v8h*)&As[arow][kbA];
    const v8h ahi = *(const v8h*)&As[arow][16 + kbA];
    const v16h af = __builtin_shufflevector(alo, ahi,
        0, 1, 2, 3, 4, 5, 6, 7, 8, 9, 10, 11, 12, 13, 14, 15);

    const int bcol = ct * 16 + l16;
    const int kbB  = half * 8;
    const v8h blo = *(const v8h*)&Bt[bcol][kbB];
    const v8h bhi = *(const v8h*)&Bt[bcol][16 + kbB];
    const v16h bf = __builtin_shufflevector(blo, bhi,
        0, 1, 2, 3, 4, 5, 6, 7, 8, 9, 10, 11, 12, 13, 14, 15);

    acc = __builtin_amdgcn_wmma_f32_16x16x32_f16(
        false, af, false, bf, (short)0, acc, false, false);
    __syncthreads();
  }

  const int col = col0 + ct * 16 + l16;
  float bv = (bias != nullptr) ? bias[col] : 0.0f;
#pragma unroll
  for (int r = 0; r < 8; ++r) {
    int m = row0 + rt * 16 + half * 8 + r;
    float v = acc[r] + bv;
    if (act) v = v > 0.0f ? v : 0.0f;
    C[(size_t)m * ldc + col] = v;
  }
}

// ===========================================================================
// Guarded WMMA GEMM for tiny shapes (M=4 y-path). Same tiling, bounds checks.
// ===========================================================================
__global__ __launch_bounds__(256) void xe_gemm_any(
    const float* __restrict__ A, int lda, int inca,
    const float* __restrict__ B, int ldb,
    const float* __restrict__ bias,
    float* __restrict__ C, int ldc,
    int M, int K, int Nn, int act)
{
  __shared__ _Float16 As[32][32];
  __shared__ _Float16 Bs[32][72];
  const int tid  = threadIdx.x;
  const int wave = tid >> 5, lane = tid & 31;
  const int rt = wave >> 2, ct = wave & 3;
  const int half = lane >> 4, l16 = lane & 15;
  const int row0 = blockIdx.x * 32, col0 = blockIdx.y * 64;
  v8f acc = {};
  for (int k0 = 0; k0 < K; k0 += 32) {
    for (int t = tid; t < 32 * 32; t += 256) {
      int r = t >> 5, c = t & 31;
      int gr = row0 + r, gk = k0 + c;
      float v = 0.0f;
      if (gr < M && gk < K) v = A[(size_t)gr * lda + (size_t)gk * inca];
      As[r][c] = (_Float16)v;
    }
    for (int t = tid; t < 32 * 64; t += 256) {
      int r = t >> 6, c = t & 63;
      int gk = k0 + r, gc = col0 + c;
      float v = 0.0f;
      if (gk < K && gc < Nn) v = B[(size_t)gk * ldb + gc];
      Bs[r][c] = (_Float16)v;
    }
    __syncthreads();
    const int arow = rt * 16 + l16;
    const int kbA  = half * 8;
    const v8h alo = *(const v8h*)&As[arow][kbA];
    const v8h ahi = *(const v8h*)&As[arow][16 + kbA];
    const v16h af = __builtin_shufflevector(alo, ahi,
        0, 1, 2, 3, 4, 5, 6, 7, 8, 9, 10, 11, 12, 13, 14, 15);
    const int bcol = ct * 16 + l16;
    const int kbB  = half * 8;
    v16h bf;
#pragma unroll
    for (int i = 0; i < 16; ++i) bf[i] = Bs[kbB + (i & 7) + ((i >> 3) << 4)][bcol];
    acc = __builtin_amdgcn_wmma_f32_16x16x32_f16(
        false, af, false, bf, (short)0, acc, false, false);
    __syncthreads();
  }
  const int col = col0 + ct * 16 + l16;
  float bv = 0.0f;
  if (bias != nullptr && col < Nn) bv = bias[col];
#pragma unroll
  for (int r = 0; r < 8; ++r) {
    int m = row0 + rt * 16 + half * 8 + r;
    if (m < M && col < Nn) {
      float v = acc[r] + bv;
      if (act) v = v > 0.0f ? v : 0.0f;
      C[(size_t)m * ldc + col] = v;
    }
  }
}

// ===========================================================================
// GEMV-8: C[r,0:8] = A[r,0:64] @ W[64,8] + b  (e_att_mul / pos_att_mul)
// ===========================================================================
__global__ __launch_bounds__(256) void xe_lin8(
    const float* __restrict__ A, const float* __restrict__ W,
    const float* __restrict__ bias, float* __restrict__ C, int M)
{
  __shared__ float sw[DEC * NHC];
  __shared__ float sb[NHC];
  for (int t = threadIdx.x; t < DEC * NHC; t += 256) sw[t] = W[t];
  if (threadIdx.x < NHC) sb[threadIdx.x] = bias[threadIdx.x];
  __syncthreads();
  int r = blockIdx.x * blockDim.x + threadIdx.x;
  if (r >= M) return;
  const float* a = A + (size_t)r * DEC;
  float acc[NHC];
#pragma unroll
  for (int h = 0; h < NHC; ++h) acc[h] = sb[h];
  for (int k = 0; k < DEC; k += 4) {
    float4 f = *(const float4*)(a + k);
#pragma unroll
    for (int h = 0; h < NHC; ++h) {
      acc[h] += f.x * sw[(k + 0) * NHC + h];
      acc[h] += f.y * sw[(k + 1) * NHC + h];
      acc[h] += f.z * sw[(k + 2) * NHC + h];
      acc[h] += f.w * sw[(k + 3) * NHC + h];
    }
  }
  float* o = C + (size_t)r * NHC;
#pragma unroll
  for (int h = 0; h < NHC; ++h) o[h] = acc[h];
}

// ===========================================================================
// Positional prep
// ===========================================================================
__global__ void xe_prep(const float* __restrict__ pos, const float* __restrict__ nm,
                        float* __restrict__ npos, float* __restrict__ normp, int total)
{
  int t = blockIdx.x * blockDim.x + threadIdx.x;
  if (t >= total) return;
  float m = nm[t];
  float x = pos[t * 3 + 0] * m;
  float y = pos[t * 3 + 1] * m;
  float z = pos[t * 3 + 2] * m;
  float nrm = sqrtf(x * x + y * y + z * z);
  float inv = 1.0f / (nrm + 1e-7f);
  npos[t * 4 + 0] = x * inv;
  npos[t * 4 + 1] = y * inv;
  npos[t * 4 + 2] = z * inv;
  npos[t * 4 + 3] = 0.0f;
  normp[t] = nrm;
}

// ===========================================================================
// dist1[b,i,j,:] = relu(Wd@[pdist,cos] + bd + norm1_i + norm2_j) * emask
// ===========================================================================
__global__ __launch_bounds__(128) void xe_dist1(
    const float* __restrict__ pos, const float* __restrict__ nm,
    const float* __restrict__ npos, const float* __restrict__ normp,
    const float* __restrict__ wd, const float* __restrict__ bd,
    const float* __restrict__ w1, const float* __restrict__ b1,
    const float* __restrict__ w2, const float* __restrict__ b2,
    float* __restrict__ dist1, int Ptot)
{
  __shared__ float s[5 * 64];
  for (int t = threadIdx.x; t < 64; t += 128) {
    s[t]       = wd[t];
    s[64 + t]  = wd[64 + t];
    s[128 + t] = bd[t] + b1[t] + b2[t];
    s[192 + t] = w1[t];
    s[256 + t] = w2[t];
  }
  __syncthreads();
  int p = blockIdx.x * 128 + threadIdx.x;
  if (p >= Ptot) return;
  int b = p / (NN * NN), rem = p % (NN * NN);
  int i = rem / NN, j = rem % NN;
  int ii = b * NN + i, jj = b * NN + j;
  float mi = nm[ii], mj = nm[jj], em = mi * mj;
  float dx = pos[ii * 3 + 0] * mi - pos[jj * 3 + 0] * mj;
  float dy = pos[ii * 3 + 1] * mi - pos[jj * 3 + 1] * mj;
  float dz = pos[ii * 3 + 2] * mi - pos[jj * 3 + 2] * mj;
  float sq = dx * dx + dy * dy + dz * dz;
  float pd = sq > 0.0f ? sqrtf(sq) : 0.0f;
  float cs = npos[ii * 4 + 0] * npos[jj * 4 + 0] +
             npos[ii * 4 + 1] * npos[jj * 4 + 1] +
             npos[ii * 4 + 2] * npos[jj * 4 + 2];
  float n1 = normp[ii], n2 = normp[jj];
  float* o = dist1 + (size_t)p * DEC;
#pragma unroll 2
  for (int c = 0; c < DEC; c += 4) {
    float4 ov;
    float v0 = s[c+0] * pd + s[64+c+0] * cs + s[128+c+0] + s[192+c+0] * n1 + s[256+c+0] * n2;
    float v1 = s[c+1] * pd + s[64+c+1] * cs + s[128+c+1] + s[192+c+1] * n1 + s[256+c+1] * n2;
    float v2 = s[c+2] * pd + s[64+c+2] * cs + s[128+c+2] + s[192+c+2] * n1 + s[256+c+2] * n2;
    float v3 = s[c+3] * pd + s[64+c+3] * cs + s[128+c+3] + s[192+c+3] * n1 + s[256+c+3] * n2;
    ov.x = (v0 > 0.0f ? v0 : 0.0f) * em;
    ov.y = (v1 > 0.0f ? v1 : 0.0f) * em;
    ov.z = (v2 > 0.0f ? v2 : 0.0f) * em;
    ov.w = (v3 > 0.0f ? v3 : 0.0f) * em;
    *(float4*)(o + c) = ov;
  }
}

// ===========================================================================
// Fused E modulation (float4): Y -> Y', Enew
// ===========================================================================
__global__ void xe_ecomb(
    const float* __restrict__ xem1, const float* __restrict__ xem2,
    const float* __restrict__ dadd, const float* __restrict__ dmul,
    const float* __restrict__ yea,  const float* __restrict__ yem,
    const float* __restrict__ nm,
    float* __restrict__ Y, float* __restrict__ Enew, size_t totalv)
{
  size_t t = (size_t)blockIdx.x * blockDim.x + threadIdx.x;
  if (t >= totalv) return;
  size_t p = t >> 4;
  int c = (int)(t & 15) << 2;
  int b = (int)(p / (NN * NN));
  int rem = (int)(p % (NN * NN));
  int i = rem / NN, j = rem % NN;
  float em = nm[b * NN + i] * nm[b * NN + j];
  const size_t e = p * DEC + c;
  float4 y0 = *(const float4*)(Y + e);
  float4 x1 = *(const float4*)(xem1 + (size_t)(b * NN + j) * DEC + c);
  float4 x2 = *(const float4*)(xem2 + (size_t)(b * NN + i) * DEC + c);
  float4 da = *(const float4*)(dadd + e);
  float4 dm = *(const float4*)(dmul + e);
  float4 ya = *(const float4*)(yea + b * DEC + c);
  float4 ym = *(const float4*)(yem + b * DEC + c);
  float4 yo, eo;
#define ECOMB1(F)                                                     \
  { float y0v = y0.F * x1.F * x2.F * em;                              \
    float y1v = (y0v + da.F + y0v * dm.F) * em;                       \
    yo.F = y1v;                                                       \
    eo.F = (y1v + ya.F + y1v * ym.F) * em; }
  ECOMB1(x) ECOMB1(y) ECOMB1(z) ECOMB1(w)
#undef ECOMB1
  *(float4*)(Y + e)    = yo;
  *(float4*)(Enew + e) = eo;
}

// ===========================================================================
// Attention logits (float4 K-dim): a[b,i,j,h] = (1/8)*sum_d Q_id K_jd aw[d,h]
// ===========================================================================
__global__ __launch_bounds__(128) void xe_att_a(
    const float* __restrict__ Q, const float* __restrict__ Kx,
    const float* __restrict__ aw, float* __restrict__ abuf, int Ptot)
{
  __shared__ float saw[DXC * NHC];
  for (int t = threadIdx.x; t < DXC * NHC; t += 128) saw[t] = aw[t];
  __syncthreads();
  int p = blockIdx.x * 128 + threadIdx.x;
  if (p >= Ptot) return;
  int b = p / (NN * NN), rem = p % (NN * NN);
  int i = rem / NN, j = rem % NN;
  const float* q = Q  + (size_t)(b * NN + i) * DXC;
  const float* k = Kx + (size_t)(b * NN + j) * DXC;
  float acc[NHC] = {0, 0, 0, 0, 0, 0, 0, 0};
  for (int d = 0; d < DXC; d += 4) {
    float4 q4 = *(const float4*)(q + d);
    float4 k4 = *(const float4*)(k + d);
    float p0 = q4.x * k4.x, p1 = q4.y * k4.y, p2 = q4.z * k4.z, p3 = q4.w * k4.w;
#pragma unroll
    for (int h = 0; h < NHC; ++h) {
      acc[h] += p0 * saw[(d + 0) * NHC + h];
      acc[h] += p1 * saw[(d + 1) * NHC + h];
      acc[h] += p2 * saw[(d + 2) * NHC + h];
      acc[h] += p3 * saw[(d + 3) * NHC + h];
    }
  }
  float* o = abuf + (size_t)p * NHC;
#pragma unroll
  for (int h = 0; h < NHC; ++h) o[h] = acc[h] * 0.125f;
}

// a = ((a*em)*(1+eatt)*(1+patt))*em ; masked-fill -1e9 where nm_j==0
__global__ void xe_attmod(float* __restrict__ abuf,
                          const float* __restrict__ eatt, const float* __restrict__ patt,
                          const float* __restrict__ nm, int Ptot)
{
  int p = blockIdx.x * blockDim.x + threadIdx.x;
  if (p >= Ptot) return;
  int b = p / (NN * NN), rem = p % (NN * NN);
  int i = rem / NN, j = rem % NN;
  float mj = nm[b * NN + j];
  float em = nm[b * NN + i] * mj;
  size_t base = (size_t)p * NHC;
#pragma unroll
  for (int h = 0; h < NHC; ++h) {
    float v = abuf[base + h] * em;
    v += eatt[base + h] * v;
    v += patt[base + h] * v;
    v *= em;
    abuf[base + h] = (mj > 0.0f) ? v : -1e9f;
  }
}

// softmax over j, in place; one thread per (b,i,h)
__global__ void xe_softmax(float* __restrict__ a, int total)
{
  int t = blockIdx.x * blockDim.x + threadIdx.x;
  if (t >= total) return;
  int h = t & 7, bi = t >> 3;
  size_t base = (size_t)bi * NN * NHC + h;
  float mx = -1e30f;
  for (int j = 0; j < NN; ++j) mx = fmaxf(mx, a[base + (size_t)j * NHC]);
  float s = 0.0f;
  for (int j = 0; j < NN; ++j) {
    float e = __expf(a[base + (size_t)j * NHC] - mx);
    a[base + (size_t)j * NHC] = e;
    s += e;
  }
  float inv = 1.0f / s;
  for (int j = 0; j < NN; ++j) a[base + (size_t)j * NHC] *= inv;
}

// ===========================================================================
// pool_stats over axis=2 (j): Z [b,i,j,C] -> out [b,i,4C]
// ===========================================================================
__global__ void xe_pool_j(const float* __restrict__ Z, const float* __restrict__ nm,
                          float* __restrict__ out, int C, int total)
{
  int t = blockIdx.x * blockDim.x + threadIdx.x;
  if (t >= total) return;
  int c = t % C, bi = t / C, b = bi / NN;
  const float* zr = Z + ((size_t)bi * NN) * C + c;
  float den = 0, sum = 0, mn = 1e30f, mx = -1e30f;
  for (int j = 0; j < NN; ++j) {
    float m = nm[b * NN + j];
    float z = zr[(size_t)j * C];
    den += m; sum += z * m;
    float im = 1.0f - m;
    mn = fminf(mn, z + 100000.0f * im);
    mx = fmaxf(mx, z - 100000.0f * im);
  }
  float mean = sum / den, var = 0;
  for (int j = 0; j < NN; ++j) {
    float m = nm[b * NN + j];
    float d = (zr[(size_t)j * C] - mean) * m;
    var += d * d;
  }
  float* o = out + (size_t)bi * 4 * C + c;
  o[0] = mean; o[C] = mn; o[2 * C] = mx; o[3 * C] = var / den;
}

// pool_stats over axis=1 (i): Z [b,i,C] -> out [b,4C]
__global__ void xe_pool_i(const float* __restrict__ Z, const float* __restrict__ nm,
                          float* __restrict__ out, int C, int total)
{
  int t = blockIdx.x * blockDim.x + threadIdx.x;
  if (t >= total) return;
  int c = t % C, b = t / C;
  float den = 0, sum = 0, mn = 1e30f, mx = -1e30f;
  for (int i = 0; i < NN; ++i) {
    float m = nm[b * NN + i];
    float z = Z[((size_t)(b * NN + i)) * C + c];
    den += m; sum += z * m;
    float im = 1.0f - m;
    mn = fminf(mn, z + 100000.0f * im);
    mx = fmaxf(mx, z - 100000.0f * im);
  }
  float mean = sum / den, var = 0;
  for (int i = 0; i < NN; ++i) {
    float m = nm[b * NN + i];
    float d = (Z[((size_t)(b * NN + i)) * C + c] - mean) * m;
    var += d * d;
  }
  float* o = out + (size_t)b * 4 * C + c;
  o[0] = mean; o[C] = mn; o[2 * C] = mx; o[3 * C] = var / den;
}

// pool_stats over axes (1,2): Z [b,i,j,C] -> out [b,4C]; one block per (b,c)
__global__ __launch_bounds__(256) void xe_pool_ij(
    const float* __restrict__ Z, const float* __restrict__ nm,
    float* __restrict__ out, int C)
{
  const int b = blockIdx.x / C, c = blockIdx.x % C;
  const int tid = threadIdx.x;
  __shared__ float rs[256], rd[256], rmn[256], rmx[256];
  float sum = 0, den = 0, mn = 1e30f, mx = -1e30f;
  const size_t base = (size_t)b * NN * NN;
  for (int p = tid; p < NN * NN; p += 256) {
    float m = nm[b * NN + p / NN] * nm[b * NN + p % NN];
    float z = Z[(base + p) * C + c];
    den += m; sum += z * m;
    float im = 1.0f - m;
    mn = fminf(mn, z + 100000.0f * im);
    mx = fmaxf(mx, z - 100000.0f * im);
  }
  rs[tid] = sum; rd[tid] = den; rmn[tid] = mn; rmx[tid] = mx;
  __syncthreads();
  for (int o = 128; o > 0; o >>= 1) {
    if (tid < o) {
      rs[tid] += rs[tid + o];
      rd[tid] += rd[tid + o];
      rmn[tid] = fminf(rmn[tid], rmn[tid + o]);
      rmx[tid] = fmaxf(rmx[tid], rmx[tid + o]);
    }
    __syncthreads();
  }
  float mean = rs[0] / rd[0];
  float denom = rd[0];
  __syncthreads();
  float var = 0;
  for (int p = tid; p < NN * NN; p += 256) {
    float m = nm[b * NN + p / NN] * nm[b * NN + p % NN];
    float d = (Z[(base + p) * C + c] - mean) * m;
    var += d * d;
  }
  rs[tid] = var;
  __syncthreads();
  for (int o = 128; o > 0; o >>= 1) {
    if (tid < o) rs[tid] += rs[tid + o];
    __syncthreads();
  }
  if (tid == 0) {
    float* o4 = out + (size_t)b * 4 * C + c;
    o4[0] = mean; o4[C] = rmn[0]; o4[2 * C] = rmx[0]; o4[3 * C] = rs[0] / denom;
  }
}

// ===========================================================================
// newX = ((wVo*xm)*(1+e_x)*(1+pos_x))*(y_x_mul+1) + y_x_add
// ===========================================================================
__global__ void xe_wvcomb(const float* __restrict__ wVo, const float* __restrict__ nm,
                          const float* __restrict__ ex,  const float* __restrict__ px,
                          const float* __restrict__ yxm, const float* __restrict__ yxa,
                          float* __restrict__ newX, int total)
{
  int t = blockIdx.x * blockDim.x + threadIdx.x;
  if (t >= total) return;
  int c = t & 255, bi = t >> 8, b = bi / NN;
  float w = wVo[t] * nm[bi];
  w += ex[t] * w;
  w += px[t] * w;
  newX[t] = w * (yxm[b * DXC + c] + 1.0f) + yxa[b * DXC + c];
}

// ===========================================================================
// LayerNorm(x + r); one block per row; out may alias x.
// ===========================================================================
__global__ __launch_bounds__(256) void xe_ln(
    const float* __restrict__ x, const float* __restrict__ r,
    const float* __restrict__ g, const float* __restrict__ bb,
    float* __restrict__ out, int C)
{
  const int row = blockIdx.x, tid = threadIdx.x;
  __shared__ float red[256];
  const float* xr = x + (size_t)row * C;
  const float* rr = r + (size_t)row * C;
  float s = 0;
  for (int c = tid; c < C; c += 256) s += xr[c] + rr[c];
  red[tid] = s;
  __syncthreads();
  for (int o = 128; o > 0; o >>= 1) {
    if (tid < o) red[tid] += red[tid + o];
    __syncthreads();
  }
  float mean = red[0] / (float)C;
  __syncthreads();
  float v = 0;
  for (int c = tid; c < C; c += 256) {
    float d = xr[c] + rr[c] - mean;
    v += d * d;
  }
  red[tid] = v;
  __syncthreads();
  for (int o = 128; o > 0; o >>= 1) {
    if (tid < o) red[tid] += red[tid + o];
    __syncthreads();
  }
  float inv = rsqrtf(red[0] / (float)C + 1e-5f);
  for (int c = tid; c < C; c += 256)
    out[(size_t)row * C + c] = (xr[c] + rr[c] - mean) * inv * g[c] + bb[c];
}

// Xr *= node_mask (float4)
__global__ void xe_maskx(float* __restrict__ Xr, const float* __restrict__ nm, int totalv)
{
  int t = blockIdx.x * blockDim.x + threadIdx.x;
  if (t >= totalv) return;
  float m = nm[t >> 6];                      // 64 float4 per row of 256
  float4 v = *(float4*)(Xr + (size_t)t * 4);
  v.x *= m; v.y *= m; v.z *= m; v.w *= m;
  *(float4*)(Xr + (size_t)t * 4) = v;
}

// Er = 0.5*(Er + Er^T) * emask (float4, pairs handled once)
__global__ void xe_sym(float* __restrict__ Er, const float* __restrict__ nm, size_t totalv)
{
  size_t t = (size_t)blockIdx.x * blockDim.x + threadIdx.x;
  if (t >= totalv) return;
  size_t p = t >> 4;
  int c = (int)(t & 15) << 2;
  int b = (int)(p / (NN * NN));
  int rem = (int)(p % (NN * NN));
  int i = rem / NN, j = rem % NN;
  if (j < i) return;
  size_t tij = p * DEC + c;
  size_t tji = (((size_t)b * NN + j) * NN + i) * DEC + c;
  float em = nm[b * NN + i] * nm[b * NN + j];
  float4 a = *(const float4*)(Er + tij);
  float4 bt = *(const float4*)(Er + tji);
  float4 v;
  v.x = 0.5f * (a.x + bt.x) * em;
  v.y = 0.5f * (a.y + bt.y) * em;
  v.z = 0.5f * (a.z + bt.z) * em;
  v.w = 0.5f * (a.w + bt.w) * em;
  *(float4*)(Er + tij) = v;
  *(float4*)(Er + tji) = v;
}

__global__ void xe_add4(const float* a, const float* b, const float* c, const float* d,
                        float* o, int n)
{
  int t = blockIdx.x * blockDim.x + threadIdx.x;
  if (t < n) o[t] = a[t] + b[t] + c[t] + d[t];
}

// ===========================================================================
// Host-side orchestration
// ===========================================================================
static inline void gemm(hipStream_t s, const float* A, int lda, int inca,
                        const float* B, int ldb, const float* bias,
                        float* C, int ldc, int M, int K, int Nn, int act)
{
  dim3 g((unsigned)((M + 31) / 32), (unsigned)((Nn + 63) / 64));
  if ((M & 31) == 0 && (Nn & 63) == 0 && (K & 31) == 0) {
    if (inca == 1)
      xe_gemm_fast<true><<<g, 256, 0, s>>>(A, lda, inca, B, ldb, bias, C, ldc, M, K, Nn, act);
    else
      xe_gemm_fast<false><<<g, 256, 0, s>>>(A, lda, inca, B, ldb, bias, C, ldc, M, K, Nn, act);
  } else {
    xe_gemm_any<<<g, 256, 0, s>>>(A, lda, inca, B, ldb, bias, C, ldc, M, K, Nn, act);
  }
}

extern "C" void kernel_launch(void* const* d_in, const int* in_sizes, int n_in,
                              void* d_out, int out_size, void* d_ws, size_t ws_size,
                              hipStream_t stream)
{
  (void)in_sizes; (void)n_in; (void)out_size; (void)ws_size;

  constexpr int    MX = BSZ * NN;                 // 768 node rows
  constexpr size_t ME = (size_t)BSZ * NN * NN;    // 147456 edge rows

  const float* X   = (const float*)d_in[0];
  const float* E   = (const float*)d_in[1];
  const float* y   = (const float*)d_in[2];
  const float* pos = (const float*)d_in[3];
  const float* nm  = (const float*)d_in[4];

  // params: dict flattened with sorted keys; within {'b','w'}: b<w; {'b','g'}: b<g
  enum {
    P_A_W = 5,
    P_DADD_B, P_DADD_W, P_DMUL_B, P_DMUL_W, P_DISTY_B, P_DISTY_W,
    P_EATT_B, P_EATT_W, P_EOUT_B, P_EOUT_W, P_EXMUL_B, P_EXMUL_W, P_EY_B, P_EY_W,
    P_INE_B, P_INE_W, P_K_B, P_K_W,
    P_LINE1_B, P_LINE1_W, P_LINE2_B, P_LINE2_W, P_LINX1_B, P_LINX1_W, P_LINX2_B, P_LINX2_W,
    P_LDIST_B, P_LDIST_W, P_LNP1_B, P_LNP1_W, P_LNP2_B, P_LNP2_W,
    P_LY1_B, P_LY1_W, P_LY2_B, P_LY2_W,
    P_NE1_B, P_NE1_G, P_NE2_B, P_NE2_G, P_NX1_B, P_NX1_G, P_NX2_B, P_NX2_G,
    P_NY1_B, P_NY1_G, P_NY2_B, P_NY2_G,
    P_OUT_B, P_OUT_W, P_PATT_B, P_PATT_W, P_PXMUL_B, P_PXMUL_W,
    P_Q_B, P_Q_W, P_V_B, P_V_W,
    P_XEM1_B, P_XEM1_W, P_XEM2_B, P_XEM2_W, P_XOUT_B, P_XOUT_W, P_XY_B, P_XY_W,
    P_YEA_B, P_YEA_W, P_YEM_B, P_YEM_W,
    P_YO1_B, P_YO1_W, P_YO2_B, P_YO2_W,
    P_YXA_B, P_YXA_W, P_YXM_B, P_YXM_W, P_YY_B, P_YY_W
  };
#define PF(i) ((const float*)d_in[(i)])

  float* Xr = (float*)d_out;                       // [4,192,256]
  float* Er = Xr + (size_t)MX * DXC;               // [4,192,192,64]
  float* yr = Er + ME * DEC;                       // [4,64]

  float* wp = (float*)d_ws;
  auto take = [&](size_t n) { float* p = wp; wp += n; return p; };
  float* dist1 = take(ME * DEC);
  float* Ybuf  = take(ME * DEC);
  float* Enew  = take(ME * DEC);
  float* t1    = take(ME * DEC);
  float* t2    = take(ME * 128);
  float* ffe   = take(ME * 128);
  float* abuf  = take(ME * NHC);
  float* eatt  = take(ME * NHC);
  float* patt  = take(ME * NHC);
  float* Qb    = take((size_t)MX * DXC);
  float* Kb    = take((size_t)MX * DXC);
  float* Vb    = take((size_t)MX * DXC);
  float* wVb   = take((size_t)MX * NHC * DXC);     // [b,i,h,d]
  float* wVo   = take((size_t)MX * DXC);
  float* xem1  = take((size_t)MX * DEC);
  float* xem2  = take((size_t)MX * DEC);
  float* poolE = take((size_t)MX * 4 * DEC);
  float* poolD = take((size_t)MX * 4 * DEC);
  float* exb   = take((size_t)MX * DXC);
  float* pxb   = take((size_t)MX * DXC);
  float* newX  = take((size_t)MX * DXC);
  float* xbuf  = take((size_t)MX * DXC);
  float* ffx   = take((size_t)MX * 2048);
  float* npos  = take((size_t)MX * 4);
  float* normp = take((size_t)MX);
  float* yea   = take(BSZ * DEC);
  float* yem   = take(BSZ * DEC);
  float* yxm   = take(BSZ * DXC);
  float* yxa   = take(BSZ * DXC);
  float* y1b   = take(BSZ * DYC);
  float* plYE  = take(BSZ * 4 * DEC);
  float* plYD  = take(BSZ * 4 * DEC);
  float* plYX  = take(BSZ * 4 * DXC);
  float* xyv   = take(BSZ * DYC);
  float* eyv   = take(BSZ * DYC);
  float* dyv   = take(BSZ * DYC);
  float* newy  = take(BSZ * DYC);
  float* yt1   = take(BSZ * DYC);
  float* yt2   = take(BSZ * DYC);
  float* ffy   = take((size_t)BSZ * 2048);

  const int THR = 256;
  auto blk = [&](size_t n) { return (unsigned)((n + THR - 1) / THR); };

  // ---- positional features + dist1 -------------------------------------
  xe_prep<<<blk(MX), THR, 0, stream>>>(pos, nm, npos, normp, MX);
  xe_dist1<<<(unsigned)((ME + 127) / 128), 128, 0, stream>>>(
      pos, nm, npos, normp,
      PF(P_LDIST_W), PF(P_LDIST_B), PF(P_LNP1_W), PF(P_LNP1_B),
      PF(P_LNP2_W), PF(P_LNP2_B), dist1, (int)ME);

  // ---- E path GEMMs (WMMA) ---------------------------------------------
  gemm(stream, X, DXC, 1, PF(P_XEM1_W), DEC, PF(P_XEM1_B), xem1, DEC, MX, DXC, DEC, 0);
  gemm(stream, X, DXC, 1, PF(P_XEM2_W), DEC, PF(P_XEM2_B), xem2, DEC, MX, DXC, DEC, 0);
  gemm(stream, E, DEC, 1, PF(P_INE_W), DEC, PF(P_INE_B), Ybuf, DEC, (int)ME, DEC, DEC, 0);
  gemm(stream, dist1, DEC, 1, PF(P_DADD_W), DEC, PF(P_DADD_B), t1, DEC, (int)ME, DEC, DEC, 0);
  gemm(stream, dist1, DEC, 1, PF(P_DMUL_W), DEC, PF(P_DMUL_B), t2, DEC, (int)ME, DEC, DEC, 0);
  gemm(stream, y, DYC, 1, PF(P_YEA_W), DEC, PF(P_YEA_B), yea, DEC, BSZ, DYC, DEC, 0);
  gemm(stream, y, DYC, 1, PF(P_YEM_W), DEC, PF(P_YEM_B), yem, DEC, BSZ, DYC, DEC, 0);
  xe_ecomb<<<blk(ME * 16), THR, 0, stream>>>(xem1, xem2, t1, t2, yea, yem, nm,
                                             Ybuf, Enew, ME * 16);

  // ---- attention --------------------------------------------------------
  xe_lin8<<<blk(ME), THR, 0, stream>>>(Enew, PF(P_EATT_W), PF(P_EATT_B), eatt, (int)ME);
  xe_lin8<<<blk(ME), THR, 0, stream>>>(dist1, PF(P_PATT_W), PF(P_PATT_B), patt, (int)ME);
  gemm(stream, X, DXC, 1, PF(P_Q_W), DXC, PF(P_Q_B), Qb, DXC, MX, DXC, DXC, 0);
  gemm(stream, X, DXC, 1, PF(P_K_W), DXC, PF(P_K_B), Kb, DXC, MX, DXC, DXC, 0);
  gemm(stream, X, DXC, 1, PF(P_V_W), DXC, PF(P_V_B), Vb, DXC, MX, DXC, DXC, 0);
  xe_att_a<<<(unsigned)((ME + 127) / 128), 128, 0, stream>>>(Qb, Kb, PF(P_A_W), abuf, (int)ME);
  xe_attmod<<<blk(ME), THR, 0, stream>>>(abuf, eatt, patt, nm, (int)ME);
  xe_softmax<<<blk((size_t)MX * NHC), THR, 0, stream>>>(abuf, MX * NHC);

  // wV[b,i,h,:] = sum_j alpha[b,i,j,h] * V[b,j,:]  (strided-A WMMA GEMMs)
  for (int b = 0; b < BSZ; ++b)
    for (int h = 0; h < NHC; ++h) {
      const float* Ap = abuf + (size_t)b * NN * NN * NHC + h;
      const float* Bp = Vb + (size_t)b * NN * DXC;
      float* Cp = wVb + ((size_t)b * NN * NHC + h) * DXC;
      gemm(stream, Ap, NN * NHC, NHC, Bp, DXC, nullptr, Cp, NHC * DXC, NN, NN, DXC, 0);
    }
  gemm(stream, wVb, NHC * DXC, 1, PF(P_OUT_W), DXC, PF(P_OUT_B), wVo, DXC,
       MX, NHC * DXC, DXC, 0);

  // ---- pooled modulation into X ----------------------------------------
  xe_pool_j<<<blk((size_t)MX * DEC), THR, 0, stream>>>(Enew, nm, poolE, DEC, MX * DEC);
  xe_pool_j<<<blk((size_t)MX * DEC), THR, 0, stream>>>(dist1, nm, poolD, DEC, MX * DEC);
  gemm(stream, poolE, 4 * DEC, 1, PF(P_EXMUL_W), DXC, PF(P_EXMUL_B), exb, DXC, MX, 4 * DEC, DXC, 0);
  gemm(stream, poolD, 4 * DEC, 1, PF(P_PXMUL_W), DXC, PF(P_PXMUL_B), pxb, DXC, MX, 4 * DEC, DXC, 0);
  gemm(stream, y, DYC, 1, PF(P_YXM_W), DXC, PF(P_YXM_B), yxm, DXC, BSZ, DYC, DXC, 0);
  gemm(stream, y, DYC, 1, PF(P_YXA_W), DXC, PF(P_YXA_B), yxa, DXC, BSZ, DYC, DXC, 0);
  xe_wvcomb<<<blk((size_t)MX * DXC), THR, 0, stream>>>(wVo, nm, exb, pxb, yxm, yxa,
                                                       newX, MX * DXC);

  // ---- X residual + FF --------------------------------------------------
  gemm(stream, newX, DXC, 1, PF(P_XOUT_W), DXC, PF(P_XOUT_B), xbuf, DXC, MX, DXC, DXC, 0);
  xe_ln<<<MX, THR, 0, stream>>>(X, xbuf, PF(P_NX1_G), PF(P_NX1_B), Xr, DXC);
  gemm(stream, Xr, DXC, 1, PF(P_LINX1_W), 2048, PF(P_LINX1_B), ffx, 2048, MX, DXC, 2048, 1);
  gemm(stream, ffx, 2048, 1, PF(P_LINX2_W), DXC, PF(P_LINX2_B), xbuf, DXC, MX, 2048, DXC, 0);
  xe_ln<<<MX, THR, 0, stream>>>(Xr, xbuf, PF(P_NX2_G), PF(P_NX2_B), Xr, DXC);
  xe_maskx<<<blk((size_t)MX * DEC), THR, 0, stream>>>(Xr, nm, MX * DEC);

  // ---- E residual + FF + symmetrize -------------------------------------
  gemm(stream, Enew, DEC, 1, PF(P_EOUT_W), DEC, PF(P_EOUT_B), t1, DEC, (int)ME, DEC, DEC, 0);
  xe_ln<<<(unsigned)ME, THR, 0, stream>>>(E, t1, PF(P_NE1_G), PF(P_NE1_B), Er, DEC);
  gemm(stream, Er, DEC, 1, PF(P_LINE1_W), 128, PF(P_LINE1_B), ffe, 128, (int)ME, DEC, 128, 1);
  gemm(stream, ffe, 128, 1, PF(P_LINE2_W), DEC, PF(P_LINE2_B), t2, DEC, (int)ME, 128, DEC, 0);
  xe_ln<<<(unsigned)ME, THR, 0, stream>>>(Er, t2, PF(P_NE2_G), PF(P_NE2_B), Er, DEC);
  xe_sym<<<blk(ME * 16), THR, 0, stream>>>(Er, nm, ME * 16);

  // ---- y path ------------------------------------------------------------
  gemm(stream, y, DYC, 1, PF(P_YY_W), DYC, PF(P_YY_B), y1b, DYC, BSZ, DYC, DYC, 0);
  xe_pool_ij<<<BSZ * DEC, THR, 0, stream>>>(Ybuf, nm, plYE, DEC);
  xe_pool_ij<<<BSZ * DEC, THR, 0, stream>>>(dist1, nm, plYD, DEC);
  xe_pool_i<<<blk((size_t)BSZ * DXC), THR, 0, stream>>>(newX, nm, plYX, DXC, BSZ * DXC);
  gemm(stream, plYX, 4 * DXC, 1, PF(P_XY_W), DYC, PF(P_XY_B), xyv, DYC, BSZ, 4 * DXC, DYC, 0);
  gemm(stream, plYE, 4 * DEC, 1, PF(P_EY_W), DYC, PF(P_EY_B), eyv, DYC, BSZ, 4 * DEC, DYC, 0);
  gemm(stream, plYD, 4 * DEC, 1, PF(P_DISTY_W), DYC, PF(P_DISTY_B), dyv, DYC, BSZ, 4 * DEC, DYC, 0);
  xe_add4<<<1, THR, 0, stream>>>(y1b, xyv, eyv, dyv, newy, BSZ * DYC);
  gemm(stream, newy, DYC, 1, PF(P_YO1_W), DYC, PF(P_YO1_B), yt1, DYC, BSZ, DYC, DYC, 1);
  gemm(stream, yt1, DYC, 1, PF(P_YO2_W), DYC, PF(P_YO2_B), yt2, DYC, BSZ, DYC, DYC, 0);
  xe_ln<<<BSZ, THR, 0, stream>>>(y, yt2, PF(P_NY1_G), PF(P_NY1_B), yr, DYC);
  gemm(stream, yr, DYC, 1, PF(P_LY1_W), 2048, PF(P_LY1_B), ffy, 2048, BSZ, DYC, 2048, 1);
  gemm(stream, ffy, 2048, 1, PF(P_LY2_W), DYC, PF(P_LY2_B), yt1, DYC, BSZ, 2048, DYC, 0);
  xe_ln<<<BSZ, THR, 0, stream>>>(yr, yt1, PF(P_NY2_G), PF(P_NY2_B), yr, DYC);
}
